// GAT_62586263437674
// MI455X (gfx1250) — matrix operand order repaired and checked by reading, not compiled, not run on hardware
//
#include <hip/hip_runtime.h>
#include <hip/hip_bf16.h>

// ---------------- problem constants (match reference) ----------------
#define N_NODES 10000
#define E_EDGES 100000
#define TOT_E   (E_EDGES + N_NODES)   // self-loops appended
#define D_IN    768
#define HEADS   4
#define CDIM    192
#define HC      768                   // HEADS * CDIM
#define D3      2304                  // 3 * D_IN
#define NEG_SLOPE 0.2f

typedef __bf16 bf16_t;
typedef __attribute__((ext_vector_type(16))) __bf16 v16bf;
typedef __attribute__((ext_vector_type(8)))  __bf16 v8bf;
typedef __attribute__((ext_vector_type(8)))  float  v8f;

// ---------------- small utility kernels ----------------
__global__ void gat_cvt_bf16(bf16_t* __restrict__ dst, const float* __restrict__ src, long n) {
    long i = (long)blockIdx.x * blockDim.x + threadIdx.x;
    if (i < n) dst[i] = (bf16_t)src[i];
}

// dst[n*K + k] = (bf16) src[k*N + n]   (weight transpose+convert, [K,N] -> [N,K])
__global__ void gat_transpose_cvt(bf16_t* __restrict__ dst, const float* __restrict__ src,
                                  int K, int Ncols) {
    long i = (long)blockIdx.x * blockDim.x + threadIdx.x;
    long tot = (long)K * Ncols;
    if (i >= tot) return;
    int n = (int)(i / K);
    int k = (int)(i % K);
    dst[i] = (bf16_t)src[(long)k * Ncols + n];
}

__global__ void gat_fill_f32(float* __restrict__ p, float v, long n) {
    long i = (long)blockIdx.x * blockDim.x + threadIdx.x;
    if (i < n) p[i] = v;
}

// ---------------- WMMA bf16 GEMM: C = [relu](A @ Bt^T + bias) ----------------
// A: [M,K] bf16 row-major. Bt: [N,K] bf16 (B transposed). Each wave computes a
// 32x64 tile = 2 (M) x 4 (N) WMMA accumulators, so per K-step of 32 we load
// 2 A-fragments + 4 B-fragments and issue 8 v_wmma -> 1.5 b128 loads / wmma.
// A fragment per ISA 16-bit A 16x32 layout; B fragment uses the same K order
// (elems 0-7 = K 8*half.., elems 8-15 = +16) of column n in Bt.
__global__ __launch_bounds__(256)
void gat_gemm_bf16(const bf16_t* __restrict__ A, const bf16_t* __restrict__ Bt,
                   const float* __restrict__ bias,
                   float* __restrict__ outF, bf16_t* __restrict__ outB,
                   int M, int Ncols, int K, int do_relu) {
    const int wave   = blockIdx.x * (blockDim.x >> 5) + (threadIdx.x >> 5);
    const int lane   = threadIdx.x & 31;
    const int tilesN = Ncols >> 6;             // 64-column supertiles
    const int tilesM = (M + 31) >> 5;          // 32-row supertiles
    const int tm = wave / tilesN;
    const int tn = wave % tilesN;
    if (tm >= tilesM) return;

    const int half = lane >> 4;                // 0: lanes 0-15, 1: lanes 16-31
    const int l15  = lane & 15;

    // A rows (clamped so the ragged last supertile never reads OOB)
    const int m0 = tm * 32 + l15;
    const int m1 = tm * 32 + 16 + l15;
    const bf16_t* Ar0 = A + (size_t)min(m0, M - 1) * K;
    const bf16_t* Ar1 = A + (size_t)min(m1, M - 1) * K;

    // B columns for the 4 N-tiles
    const bf16_t* Br0 = Bt + (size_t)(tn * 64 +  0 + l15) * K;
    const bf16_t* Br1 = Bt + (size_t)(tn * 64 + 16 + l15) * K;
    const bf16_t* Br2 = Bt + (size_t)(tn * 64 + 32 + l15) * K;
    const bf16_t* Br3 = Bt + (size_t)(tn * 64 + 48 + l15) * K;

    v8f acc[2][4] = {};
    for (int k0 = 0; k0 < K; k0 += 32) {
        // A fragments: elems 0-7 = K k0+8*half.. ; elems 8-15 = +16
        v8bf a0lo = *(const v8bf*)(Ar0 + k0 + half * 8);
        v8bf a0hi = *(const v8bf*)(Ar0 + k0 + 16 + half * 8);
        v8bf a1lo = *(const v8bf*)(Ar1 + k0 + half * 8);
        v8bf a1hi = *(const v8bf*)(Ar1 + k0 + 16 + half * 8);
        v16bf a0 = __builtin_shufflevector(a0lo, a0hi,
                        0,1,2,3,4,5,6,7,8,9,10,11,12,13,14,15);
        v16bf a1 = __builtin_shufflevector(a1lo, a1hi,
                        0,1,2,3,4,5,6,7,8,9,10,11,12,13,14,15);
        // B fragments: same K order as A (elems 0-7 = K k0+8*half.., elems 8-15 = +16)
        v16bf b0 = __builtin_shufflevector(*(const v8bf*)(Br0 + k0 + half * 8), *(const v8bf*)(Br0 + k0 + 16 + half * 8), 0,1,2,3,4,5,6,7,8,9,10,11,12,13,14,15);
        v16bf b1 = __builtin_shufflevector(*(const v8bf*)(Br1 + k0 + half * 8), *(const v8bf*)(Br1 + k0 + 16 + half * 8), 0,1,2,3,4,5,6,7,8,9,10,11,12,13,14,15);
        v16bf b2 = __builtin_shufflevector(*(const v8bf*)(Br2 + k0 + half * 8), *(const v8bf*)(Br2 + k0 + 16 + half * 8), 0,1,2,3,4,5,6,7,8,9,10,11,12,13,14,15);
        v16bf b3 = __builtin_shufflevector(*(const v8bf*)(Br3 + k0 + half * 8), *(const v8bf*)(Br3 + k0 + 16 + half * 8), 0,1,2,3,4,5,6,7,8,9,10,11,12,13,14,15);

        acc[0][0] = __builtin_amdgcn_wmma_f32_16x16x32_bf16(false, a0, false, b0, (short)0, acc[0][0], false, false);
        acc[0][1] = __builtin_amdgcn_wmma_f32_16x16x32_bf16(false, a0, false, b1, (short)0, acc[0][1], false, false);
        acc[0][2] = __builtin_amdgcn_wmma_f32_16x16x32_bf16(false, a0, false, b2, (short)0, acc[0][2], false, false);
        acc[0][3] = __builtin_amdgcn_wmma_f32_16x16x32_bf16(false, a0, false, b3, (short)0, acc[0][3], false, false);
        acc[1][0] = __builtin_amdgcn_wmma_f32_16x16x32_bf16(false, a1, false, b0, (short)0, acc[1][0], false, false);
        acc[1][1] = __builtin_amdgcn_wmma_f32_16x16x32_bf16(false, a1, false, b1, (short)0, acc[1][1], false, false);
        acc[1][2] = __builtin_amdgcn_wmma_f32_16x16x32_bf16(false, a1, false, b2, (short)0, acc[1][2], false, false);
        acc[1][3] = __builtin_amdgcn_wmma_f32_16x16x32_bf16(false, a1, false, b3, (short)0, acc[1][3], false, false);
    }

    #pragma unroll
    for (int j = 0; j < 4; ++j) {
        const int n = tn * 64 + j * 16 + l15;
        const float bb = bias ? bias[n] : 0.0f;
        #pragma unroll
        for (int i = 0; i < 2; ++i) {
            #pragma unroll
            for (int r = 0; r < 8; ++r) {
                const int row = tm * 32 + i * 16 + half * 8 + r;   // C/D VGPR layout
                if (row < M) {
                    float v = acc[i][j][r] + bb;
                    if (do_relu) v = fmaxf(v, 0.0f);
                    const size_t idx = (size_t)row * Ncols + n;
                    if (outF) outF[idx] = v;
                    if (outB) outB[idx] = (bf16_t)v;
                }
            }
        }
    }
}

// ---------------- attention scores: as_n / ad_n, wave per (node, head) ----------------
__global__ __launch_bounds__(256)
void gat_attn_scores(const float* __restrict__ h,
                     const float* __restrict__ a_src, const float* __restrict__ a_dst,
                     float* __restrict__ asn, float* __restrict__ adn) {
    const int wave = blockIdx.x * (blockDim.x >> 5) + (threadIdx.x >> 5);
    const int lane = threadIdx.x & 31;
    if (wave >= N_NODES * HEADS) return;
    const int node = wave / HEADS;
    const int hh   = wave % HEADS;
    const float* hp = h + (size_t)node * HC + hh * CDIM;
    const float* as = a_src + hh * CDIM;
    const float* ad = a_dst + hh * CDIM;
    float s1 = 0.f, s2 = 0.f;
    for (int c = lane; c < CDIM; c += 32) {
        const float v = hp[c];
        s1 += v * as[c];
        s2 += v * ad[c];
    }
    #pragma unroll
    for (int off = 16; off > 0; off >>= 1) {
        s1 += __shfl_down(s1, off, 32);
        s2 += __shfl_down(s2, off, 32);
    }
    if (lane == 0) { asn[wave] = s1; adn[wave] = s2; }
}

// ---------------- edge helpers ----------------
__device__ inline void gat_edge_sd(const int* __restrict__ ei, int eid, int& s, int& d) {
    if (eid < E_EDGES) { s = ei[eid]; d = ei[E_EDGES + eid]; }
    else               { s = d = eid - E_EDGES; }               // self loop
}

__device__ inline void gat_atomic_max_f32(float* addr, float val) {
    unsigned int* ua = (unsigned int*)addr;
    unsigned int old = __float_as_uint(*addr);
    while (__uint_as_float(old) < val) {
        unsigned int assumed = old;
        old = atomicCAS(ua, assumed, __float_as_uint(val));
        if (old == assumed) break;
    }
}

// e = leaky_relu(as[src] + ad[dst]); segment max into m[dst]
__global__ void gat_edge_logits(const float* __restrict__ asn, const float* __restrict__ adn,
                                const int* __restrict__ ei,
                                float* __restrict__ e, float* __restrict__ m) {
    long i = (long)blockIdx.x * blockDim.x + threadIdx.x;
    if (i >= (long)TOT_E * HEADS) return;
    const int eid = (int)(i / HEADS);
    const int hh  = (int)(i % HEADS);
    int s, d; gat_edge_sd(ei, eid, s, d);
    float v = asn[(size_t)s * HEADS + hh] + adn[(size_t)d * HEADS + hh];
    v = (v > 0.f) ? v : NEG_SLOPE * v;
    e[i] = v;
    gat_atomic_max_f32(&m[(size_t)d * HEADS + hh], v);
}

// p = exp(e - m[dst]) in-place; segment sum into ssum[dst]
__global__ void gat_edge_exp(float* __restrict__ e, const float* __restrict__ m,
                             float* __restrict__ ssum, const int* __restrict__ ei) {
    long i = (long)blockIdx.x * blockDim.x + threadIdx.x;
    if (i >= (long)TOT_E * HEADS) return;
    const int eid = (int)(i / HEADS);
    const int hh  = (int)(i % HEADS);
    int s, d; gat_edge_sd(ei, eid, s, d);
    (void)s;
    const float v = __expf(e[i] - m[(size_t)d * HEADS + hh]);
    e[i] = v;
    atomicAdd(&ssum[(size_t)d * HEADS + hh], v);
}

// acc[dst, :] += alpha * h[src, :]   (block per edge, 768 channels)
__global__ __launch_bounds__(256)
void gat_edge_aggregate(const float* __restrict__ p, const float* __restrict__ ssum,
                        const float* __restrict__ h, const int* __restrict__ ei,
                        float* __restrict__ acc) {
    const int eid = blockIdx.x;
    int s, d; gat_edge_sd(ei, eid, s, d);
    const float* hs = h   + (size_t)s * HC;
    float*       ac = acc + (size_t)d * HC;
    for (int c = threadIdx.x; c < HC; c += blockDim.x) {
        const int hh = c / CDIM;
        const float alpha = p[(size_t)eid * HEADS + hh] /
                            (ssum[(size_t)d * HEADS + hh] + 1e-16f);
        atomicAdd(&ac[c], alpha * hs[c]);
    }
}

// emb_bf16[node, layer*768 + c] = (bf16) relu(acc[node,c] + bias[c])
__global__ void gat_finalize_layer(const float* __restrict__ acc, const float* __restrict__ bias,
                                   bf16_t* __restrict__ emb, int layer) {
    long i = (long)blockIdx.x * blockDim.x + threadIdx.x;
    if (i >= (long)N_NODES * HC) return;
    const int node = (int)(i / HC);
    const int c    = (int)(i % HC);
    const float v  = fmaxf(acc[i] + bias[c], 0.f);
    emb[(size_t)node * D3 + (size_t)layer * HC + c] = (bf16_t)v;
}

// ---------------- host side ----------------
static inline char* gat_carve(char*& p, size_t bytes) {
    char* r = p;
    p += (bytes + 255) & ~(size_t)255;
    return r;
}
static inline int gat_cdiv(long a, long b) { return (int)((a + b - 1) / b); }

extern "C" void kernel_launch(void* const* d_in, const int* in_sizes, int n_in,
                              void* d_out, int out_size, void* d_ws, size_t ws_size,
                              hipStream_t stream) {
    (void)in_sizes; (void)n_in; (void)out_size; (void)ws_size;

    const float* x      = (const float*)d_in[0];
    const int*   eidx   = (const int*)  d_in[1];
    // d_in[2] = batch, d_in[3] = edge_attr: unused by reference math
    const float* W_ah   = (const float*)d_in[4];
    const float* b_ah   = (const float*)d_in[5];
    const float* W_l[3]    = { (const float*)d_in[6],  (const float*)d_in[10], (const float*)d_in[14] };
    const float* a_src_l[3]= { (const float*)d_in[7],  (const float*)d_in[11], (const float*)d_in[15] };
    const float* a_dst_l[3]= { (const float*)d_in[8],  (const float*)d_in[12], (const float*)d_in[16] };
    const float* b_l[3]    = { (const float*)d_in[9],  (const float*)d_in[13], (const float*)d_in[17] };
    const float* W_out  = (const float*)d_in[18];
    const float* b_out  = (const float*)d_in[19];
    float* out = (float*)d_out;

    // ---- workspace carve ----
    char* p = (char*)d_ws;
    bf16_t* xb     = (bf16_t*)gat_carve(p, (size_t)N_NODES * D_IN * 2);   // x in bf16
    bf16_t* Wt_ah  = (bf16_t*)gat_carve(p, (size_t)D_IN * HC * 2);        // [N,K] bf16
    bf16_t* Wt0    = (bf16_t*)gat_carve(p, (size_t)D_IN * HC * 2);
    bf16_t* Wt1    = (bf16_t*)gat_carve(p, (size_t)D_IN * HC * 2);
    bf16_t* Wt2    = (bf16_t*)gat_carve(p, (size_t)D_IN * HC * 2);
    bf16_t* Wt_out = (bf16_t*)gat_carve(p, (size_t)D3 * D_IN * 2);
    bf16_t* x1b    = (bf16_t*)gat_carve(p, (size_t)N_NODES * D_IN * 2);   // relu(x@W_ah+b) bf16
    float*  hbuf   = (float*) gat_carve(p, (size_t)N_NODES * HC * 4);     // per-layer h
    float*  accbuf = (float*) gat_carve(p, (size_t)N_NODES * HC * 4);     // per-layer aggregation
    bf16_t* embb   = (bf16_t*)gat_carve(p, (size_t)N_NODES * D3 * 2);     // concat(3 layers) bf16
    float*  asn    = (float*) gat_carve(p, (size_t)N_NODES * HEADS * 4);
    float*  adn    = (float*) gat_carve(p, (size_t)N_NODES * HEADS * 4);
    float*  ebuf   = (float*) gat_carve(p, (size_t)TOT_E * HEADS * 4);    // e then p
    float*  mbuf   = (float*) gat_carve(p, (size_t)N_NODES * HEADS * 4);
    float*  sbuf   = (float*) gat_carve(p, (size_t)N_NODES * HEADS * 4);
    bf16_t* Wt_l[3] = { Wt0, Wt1, Wt2 };

    const int T = 256;

    // ---- convert inputs to bf16 / transposed-bf16 weights ----
    gat_cvt_bf16<<<gat_cdiv((long)N_NODES * D_IN, T), T, 0, stream>>>(xb, x, (long)N_NODES * D_IN);
    gat_transpose_cvt<<<gat_cdiv((long)D_IN * HC, T), T, 0, stream>>>(Wt_ah, W_ah, D_IN, HC);
    for (int l = 0; l < 3; ++l)
        gat_transpose_cvt<<<gat_cdiv((long)D_IN * HC, T), T, 0, stream>>>(Wt_l[l], W_l[l], D_IN, HC);
    gat_transpose_cvt<<<gat_cdiv((long)D3 * D_IN, T), T, 0, stream>>>(Wt_out, W_out, D3, D_IN);

    // GEMM grid: waves = ceil(M/32) * (N/64), 8 waves per block
    const int gemm_waves = gat_cdiv(N_NODES, 32) * (HC / 64);
    const int gemm_blocks = gat_cdiv(gemm_waves, 8);

    // ---- x1 = relu(x @ W_ah + b_ah), stored bf16 for chained GEMMs ----
    gat_gemm_bf16<<<gemm_blocks, T, 0, stream>>>(
        xb, Wt_ah, b_ah, nullptr, x1b, N_NODES, HC, D_IN, 1);

    // ---- three GAT layers, all consuming x1 ----
    for (int l = 0; l < 3; ++l) {
        gat_gemm_bf16<<<gemm_blocks, T, 0, stream>>>(
            x1b, Wt_l[l], nullptr, hbuf, nullptr, N_NODES, HC, D_IN, 0);

        gat_attn_scores<<<gat_cdiv((long)N_NODES * HEADS, 8), T, 0, stream>>>(
            hbuf, a_src_l[l], a_dst_l[l], asn, adn);

        gat_fill_f32<<<gat_cdiv((long)N_NODES * HEADS, T), T, 0, stream>>>(
            mbuf, -3.4e38f, (long)N_NODES * HEADS);
        gat_fill_f32<<<gat_cdiv((long)N_NODES * HEADS, T), T, 0, stream>>>(
            sbuf, 0.f, (long)N_NODES * HEADS);
        gat_fill_f32<<<gat_cdiv((long)N_NODES * HC, T), T, 0, stream>>>(
            accbuf, 0.f, (long)N_NODES * HC);

        gat_edge_logits<<<gat_cdiv((long)TOT_E * HEADS, T), T, 0, stream>>>(
            asn, adn, eidx, ebuf, mbuf);
        gat_edge_exp<<<gat_cdiv((long)TOT_E * HEADS, T), T, 0, stream>>>(
            ebuf, mbuf, sbuf, eidx);
        gat_edge_aggregate<<<TOT_E, T, 0, stream>>>(ebuf, sbuf, hbuf, eidx, accbuf);

        gat_finalize_layer<<<gat_cdiv((long)N_NODES * HC, T), T, 0, stream>>>(
            accbuf, b_l[l], embb, l);
    }

    // ---- out = relu(concat(xs) @ W_out + b_out) ----
    gat_gemm_bf16<<<gemm_blocks, T, 0, stream>>>(
        embb, Wt_out, b_out, out, nullptr, N_NODES, D_IN, D3, 1);
}
